// GraphAttnEncoder_6150393168616
// MI455X (gfx1250) — matrix operand order repaired and checked by reading, not compiled, not run on hardware
//
#include <hip/hip_runtime.h>
#include <hip/hip_bf16.h>

#define NNODES 50000
#define NPAD   50016      // padded to multiple of 32 for 32-row GEMM tiles
#define NEDGES 800000
#define FDIM   256
#define D2     512
#define D4     1024
#define NHEAD  4
#define HIDD   64

typedef __attribute__((ext_vector_type(16))) __bf16 bf16x16;
typedef __attribute__((ext_vector_type(8)))  float  f32x8;

__device__ __forceinline__ unsigned short f2bf(float f) {
    unsigned u = __float_as_uint(f);
    unsigned r = (u + 0x7FFFu + ((u >> 16) & 1u)) >> 16;   // round-nearest-even
    return (unsigned short)r;
}

__device__ __forceinline__ bf16x16 mkfrag(uint4 q0, uint4 q1) {
    union { bf16x16 v; uint4 q[2]; } u;
    u.q[0] = q0; u.q[1] = q1;
    return u.v;
}

// ---------------- generic f32 -> bf16 cast ----------------
__global__ __launch_bounds__(256) void cast_bf16_k(const float* __restrict__ s,
                                                   unsigned short* __restrict__ d, int n) {
    int t = blockIdx.x * 256 + threadIdx.x;
    if (t < n) d[t] = f2bf(s[t]);
}

// ---------------- build x = concat(features, emb[labels]) in bf16 ----------------
__global__ __launch_bounds__(256) void build_x_k(const float* __restrict__ feats,
                                                 const float* __restrict__ emb,
                                                 const int* __restrict__ labels,
                                                 unsigned short* __restrict__ X) {
    int t = blockIdx.x * 256 + threadIdx.x;
    if (t >= NNODES * D2) return;
    int n = t >> 9, j = t & 511;
    float v = (j < FDIM) ? feats[(size_t)n * FDIM + j]
                         : emb[(size_t)labels[n] * FDIM + (j - FDIM)];
    X[t] = f2bf(v);
}

// ---------------- WMMA bf16 GEMM: C[M,N] = A[M,K] * B[N,K]^T (+bias) ----------------
// one wave computes a 32x64 tile; M%32==0, N%64==0, K%32==0, K>=32
// double-buffered fragment loads so VMEM overlaps the WMMA pipe
__global__ __launch_bounds__(256) void gemm_bf16_k(const unsigned short* __restrict__ A,
                                                   const unsigned short* __restrict__ B,
                                                   const float* __restrict__ bias,
                                                   float* __restrict__ Cf,
                                                   unsigned short* __restrict__ Cb,
                                                   int M, int N, int K) {
    int gwave = (blockIdx.x * 256 + threadIdx.x) >> 5;
    int lane  = threadIdx.x & 31;
    int ntn   = N >> 6;
    int total = (M >> 5) * ntn;
    if (gwave >= total) return;
    int tileM = (gwave / ntn) << 5;
    int tileN = (gwave % ntn) << 6;

    int hi = lane >> 4;      // half-wave select
    int rw = lane & 15;

    f32x8 acc[2][4];
#pragma unroll
    for (int i = 0; i < 2; ++i)
#pragma unroll
        for (int j = 0; j < 4; ++j) acc[i][j] = f32x8{0};

    // A frag: lane holds row; K chunks [k0+8*hi, +8) and [k0+16+8*hi, +8)
    const unsigned short* a0 = A + (size_t)(tileM + rw) * K + (hi ? 8 : 0);
    const unsigned short* a1 = a0 + (size_t)16 * K;
    // B frag: lane holds col (= row of W); K chunks [k0+8*hi, +8) and [k0+16+8*hi, +8)
    const unsigned short* bb = B + (size_t)(tileN + rw) * K + (hi ? 8 : 0);

    uint4 aq[2][2], bq[4][2];
    aq[0][0] = *(const uint4*)(a0);      aq[0][1] = *(const uint4*)(a0 + 16);
    aq[1][0] = *(const uint4*)(a1);      aq[1][1] = *(const uint4*)(a1 + 16);
#pragma unroll
    for (int j = 0; j < 4; ++j) {
        const unsigned short* bp = bb + (size_t)(j * 16) * K;
        bq[j][0] = *(const uint4*)(bp);  bq[j][1] = *(const uint4*)(bp + 16);
    }

    for (int k0 = 32; k0 < K; k0 += 32) {
        uint4 naq[2][2], nbq[4][2];
        // issue next-iteration loads first so they overlap this iteration's WMMAs
        naq[0][0] = *(const uint4*)(a0 + k0); naq[0][1] = *(const uint4*)(a0 + k0 + 16);
        naq[1][0] = *(const uint4*)(a1 + k0); naq[1][1] = *(const uint4*)(a1 + k0 + 16);
#pragma unroll
        for (int j = 0; j < 4; ++j) {
            const unsigned short* bp = bb + (size_t)(j * 16) * K + k0;
            nbq[j][0] = *(const uint4*)(bp); nbq[j][1] = *(const uint4*)(bp + 16);
        }
        bf16x16 av0 = mkfrag(aq[0][0], aq[0][1]);
        bf16x16 av1 = mkfrag(aq[1][0], aq[1][1]);
#pragma unroll
        for (int j = 0; j < 4; ++j) {
            bf16x16 bv = mkfrag(bq[j][0], bq[j][1]);
            acc[0][j] = __builtin_amdgcn_wmma_f32_16x16x32_bf16(
                false, av0, false, bv, (short)0, acc[0][j], false, false);
            acc[1][j] = __builtin_amdgcn_wmma_f32_16x16x32_bf16(
                false, av1, false, bv, (short)0, acc[1][j], false, false);
        }
#pragma unroll
        for (int i = 0; i < 2; ++i) { aq[i][0] = naq[i][0]; aq[i][1] = naq[i][1]; }
#pragma unroll
        for (int j = 0; j < 4; ++j) { bq[j][0] = nbq[j][0]; bq[j][1] = nbq[j][1]; }
    }
    {   // epilogue K-step
        bf16x16 av0 = mkfrag(aq[0][0], aq[0][1]);
        bf16x16 av1 = mkfrag(aq[1][0], aq[1][1]);
#pragma unroll
        for (int j = 0; j < 4; ++j) {
            bf16x16 bv = mkfrag(bq[j][0], bq[j][1]);
            acc[0][j] = __builtin_amdgcn_wmma_f32_16x16x32_bf16(
                false, av0, false, bv, (short)0, acc[0][j], false, false);
            acc[1][j] = __builtin_amdgcn_wmma_f32_16x16x32_bf16(
                false, av1, false, bv, (short)0, acc[1][j], false, false);
        }
    }

    int rowb = tileM + (hi ? 8 : 0);
#pragma unroll
    for (int i = 0; i < 2; ++i) {
#pragma unroll
        for (int j = 0; j < 4; ++j) {
            int col = tileN + j * 16 + rw;
            float bv = bias ? bias[col] : 0.0f;
#pragma unroll
            for (int r = 0; r < 8; ++r) {
                float v = acc[i][j][r] + bv;
                size_t idx = (size_t)(rowb + i * 16 + r) * N + col;
                if (Cb) Cb[idx] = f2bf(v);
                else    Cf[idx] = v;
            }
        }
    }
}

// ---------------- LayerNorm + ReLU -> bf16 (row = 1024) ----------------
__global__ __launch_bounds__(256) void ln_relu_k(const float* __restrict__ X,
                                                 const float* __restrict__ g,
                                                 const float* __restrict__ b,
                                                 unsigned short* __restrict__ Y) {
    __shared__ float red[256];
    int row = blockIdx.x;
    const float* x = X + (size_t)row * D4;
    float s = 0.f, s2 = 0.f;
    for (int i = threadIdx.x; i < D4; i += 256) { float v = x[i]; s += v; s2 += v * v; }
    red[threadIdx.x] = s; __syncthreads();
    for (int o = 128; o > 0; o >>= 1) { if (threadIdx.x < o) red[threadIdx.x] += red[threadIdx.x + o]; __syncthreads(); }
    float mean = red[0] / D4; __syncthreads();
    red[threadIdx.x] = s2; __syncthreads();
    for (int o = 128; o > 0; o >>= 1) { if (threadIdx.x < o) red[threadIdx.x] += red[threadIdx.x + o]; __syncthreads(); }
    float var = red[0] / D4 - mean * mean;
    float rstd = rsqrtf(var + 1e-5f);
    unsigned short* y = Y + (size_t)row * D4;
    for (int i = threadIdx.x; i < D4; i += 256) {
        float v = (x[i] - mean) * rstd * g[i] + b[i];
        y[i] = f2bf(fmaxf(v, 0.f));
    }
}

// ---------------- el/er = <feat[n,h,:], al/ar[h,:]> ----------------
__global__ __launch_bounds__(256) void attn_scores_k(const float* __restrict__ feat,
                                                     const float* __restrict__ al,
                                                     const float* __restrict__ ar,
                                                     float* __restrict__ el,
                                                     float* __restrict__ er) {
    int t = blockIdx.x * 256 + threadIdx.x;       // n*4 + h
    if (t >= NNODES * NHEAD) return;
    int h = t & 3;
    const float4* f4 = (const float4*)(feat + (size_t)(t >> 2) * FDIM + h * HIDD);
    const float4* A4 = (const float4*)(al + h * HIDD);
    const float4* R4 = (const float4*)(ar + h * HIDD);
    float sl = 0.f, sr = 0.f;
#pragma unroll 4
    for (int d = 0; d < HIDD / 4; ++d) {
        float4 v = f4[d], a = A4[d], r = R4[d];
        sl += v.x * a.x + v.y * a.y + v.z * a.z + v.w * a.w;
        sr += v.x * r.x + v.y * r.y + v.z * r.z + v.w * r.w;
    }
    el[t] = sl; er[t] = sr;
}

// ---------------- e = leaky_relu(el[src]+er[dst]); segment max via encoded atomicMax ----------------
__global__ __launch_bounds__(256) void edge_logits_k(const float* __restrict__ el,
                                                     const float* __restrict__ er,
                                                     const int* __restrict__ src,
                                                     const int* __restrict__ dst,
                                                     float* __restrict__ e,
                                                     unsigned* __restrict__ menc) {
    int t = blockIdx.x * 256 + threadIdx.x;       // edge*4 + h
    if (t >= NEDGES * NHEAD) return;
    int ed = t >> 2, h = t & 3;
    float v = el[src[ed] * 4 + h] + er[dst[ed] * 4 + h];
    v = (v > 0.f) ? v : 0.2f * v;
    e[t] = v;
    unsigned bits = __float_as_uint(v);
    unsigned enc = (bits & 0x80000000u) ? ~bits : (bits | 0x80000000u);
    atomicMax(&menc[dst[ed] * 4 + h], enc);
}

__global__ __launch_bounds__(256) void decode_m_k(const unsigned* __restrict__ menc,
                                                  float* __restrict__ m) {
    int t = blockIdx.x * 256 + threadIdx.x;
    if (t >= NNODES * NHEAD) return;
    unsigned u = menc[t];
    float v = 0.f;                                 // no incoming edges -> 0 (isfinite fixup)
    if (u != 0u) {
        unsigned bits = (u & 0x80000000u) ? (u & 0x7FFFFFFFu) : ~u;
        v = __uint_as_float(bits);
    }
    m[t] = v;
}

// ---------------- ex = exp(e - m[dst]); den[dst] += ex ----------------
__global__ __launch_bounds__(256) void edge_exp_k(float* __restrict__ e,
                                                  const float* __restrict__ m,
                                                  const int* __restrict__ dst,
                                                  float* __restrict__ den) {
    int t = blockIdx.x * 256 + threadIdx.x;
    if (t >= NEDGES * NHEAD) return;
    int ed = t >> 2, h = t & 3;
    int d = dst[ed];
    float ex = __expf(e[t] - m[d * 4 + h]);
    e[t] = ex;
    atomicAdd(&den[d * 4 + h], ex);
}

// ---------------- one wave per edge: agg[dst] += (ex/den) * feat[src] ----------------
__global__ __launch_bounds__(256) void edge_agg_k(const float* __restrict__ ex,
                                                  const float* __restrict__ den,
                                                  const int* __restrict__ src,
                                                  const int* __restrict__ dst,
                                                  const float* __restrict__ feat,
                                                  float* __restrict__ agg) {
    int wave = (blockIdx.x * 256 + threadIdx.x) >> 5;
    int lane = threadIdx.x & 31;
    if (wave >= NEDGES) return;
    int s = src[wave], d = dst[wave];
    int h = lane >> 3;                              // (lane*8)/64
    float a = ex[(size_t)wave * 4 + h] / fmaxf(den[d * 4 + h], 1e-9f);
    const float4* fp = (const float4*)(feat + (size_t)s * FDIM + lane * 8);
    float4 v0 = fp[0], v1 = fp[1];
    float* ap = agg + (size_t)d * FDIM + lane * 8;
    atomicAdd(ap + 0, a * v0.x); atomicAdd(ap + 1, a * v0.y);
    atomicAdd(ap + 2, a * v0.z); atomicAdd(ap + 3, a * v0.w);
    atomicAdd(ap + 4, a * v1.x); atomicAdd(ap + 5, a * v1.y);
    atomicAdd(ap + 6, a * v1.z); atomicAdd(ap + 7, a * v1.w);
}

// ---------------- out = relu(agg + bias); optional f32 and bf16 outputs ----------------
__global__ __launch_bounds__(256) void finalize_k(const float* __restrict__ agg,
                                                  const float* __restrict__ bias,
                                                  float* __restrict__ outf,
                                                  unsigned short* __restrict__ outb) {
    int t = blockIdx.x * 256 + threadIdx.x;
    if (t >= NNODES * FDIM) return;
    float v = fmaxf(agg[t] + bias[t & 255], 0.f);
    if (outf) outf[t] = v;
    if (outb) outb[t] = f2bf(v);
}

static inline int cdiv(long long a, long long b) { return (int)((a + b - 1) / b); }

extern "C" void kernel_launch(void* const* d_in, const int* in_sizes, int n_in,
                              void* d_out, int out_size, void* d_ws, size_t ws_size,
                              hipStream_t stream) {
    const float* features = (const float*)d_in[0];
    const int*   labels   = (const int*)d_in[1];
    const int*   src      = (const int*)d_in[2];
    const int*   dst      = (const int*)d_in[3];
    const float* emb      = (const float*)d_in[4];
    const float* le_W1    = (const float*)d_in[5];
    const float* le_b1    = (const float*)d_in[6];
    const float* ln_g     = (const float*)d_in[7];
    const float* ln_b     = (const float*)d_in[8];
    const float* le_W2    = (const float*)d_in[9];
    const float* le_b2    = (const float*)d_in[10];
    const float* W1       = (const float*)d_in[11];
    const float* al1      = (const float*)d_in[12];
    const float* ar1      = (const float*)d_in[13];
    const float* b1       = (const float*)d_in[14];
    const float* W2       = (const float*)d_in[15];
    const float* al2      = (const float*)d_in[16];
    const float* ar2      = (const float*)d_in[17];
    const float* b2       = (const float*)d_in[18];

    char* W = (char*)d_ws;
    // big region (205 MB): Y1 [NPAD,1024] pre-LN activations; later reused: feat + agg
    float*          Y1   = (float*)(W + 0);
    float*          feat = (float*)(W + 0);                    // [NPAD,256] f32
    float*          agg  = (float*)(W + 51216384);             // [NNODES,256] f32
    // X region: X_bf [NPAD,512] first, X2_bf [NPAD,1024] after (X_bf dead)
    unsigned short* Xbf  = (unsigned short*)(W + 204865536);
    unsigned short* X2bf = (unsigned short*)(W + 204865536);
    unsigned short* wb1  = (unsigned short*)(W + 307298304);   // le_W1 bf16
    unsigned short* wb2  = (unsigned short*)(W + 308346880);   // le_W2 bf16
    unsigned short* wg1  = (unsigned short*)(W + 308871168);   // W1 bf16
    unsigned short* wg2  = (unsigned short*)(W + 309002240);   // W2 bf16
    unsigned short* Hbf  = (unsigned short*)(W + 309133312);   // layer-1 input bf16 [NPAD,256]
    unsigned short* H1bf = (unsigned short*)(W + 334741504);   // layer-1 output bf16 [NPAD,256]
    float*          el   = (float*)(W + 360349696);
    float*          er   = (float*)(W + 361149696);
    unsigned*       menc = (unsigned*)(W + 361949696);
    float*          m    = (float*)(W + 362749696);
    float*          den  = (float*)(W + 363549696);
    float*          ebuf = (float*)(W + 364349696);            // e then ex, [E,4]

    // ---- weight conversion ----
    cast_bf16_k<<<cdiv(D4 * D2, 256), 256, 0, stream>>>(le_W1, wb1, D4 * D2);
    cast_bf16_k<<<cdiv(FDIM * D4, 256), 256, 0, stream>>>(le_W2, wb2, FDIM * D4);
    cast_bf16_k<<<cdiv(FDIM * FDIM, 256), 256, 0, stream>>>(W1, wg1, FDIM * FDIM);
    cast_bf16_k<<<cdiv(FDIM * FDIM, 256), 256, 0, stream>>>(W2, wg2, FDIM * FDIM);

    // ---- label-embed MLP ----
    build_x_k<<<cdiv((long long)NNODES * D2, 256), 256, 0, stream>>>(features, emb, labels, Xbf);
    {   // Y1 = X @ le_W1^T + b1  [NPAD,1024]
        long long waves = (NPAD / 32) * (D4 / 64);
        gemm_bf16_k<<<cdiv(waves * 32, 256), 256, 0, stream>>>(Xbf, wb1, le_b1, Y1, nullptr,
                                                               NPAD, D4, D2);
    }
    ln_relu_k<<<NNODES, 256, 0, stream>>>(Y1, ln_g, ln_b, X2bf);
    {   // Hbf = (X2 @ le_W2^T + b2) -> bf16  [NPAD,256]
        long long waves = (NPAD / 32) * (FDIM / 64);
        gemm_bf16_k<<<cdiv(waves * 32, 256), 256, 0, stream>>>(X2bf, wb2, le_b2, nullptr, Hbf,
                                                               NPAD, FDIM, D4);
    }

    // ---- two GAT layers ----
    const unsigned short* layer_in[2] = {Hbf, H1bf};
    const unsigned short* layer_W[2]  = {wg1, wg2};
    const float* layer_al[2] = {al1, al2};
    const float* layer_ar[2] = {ar1, ar2};
    const float* layer_b[2]  = {b1, b2};
    float*          layer_outf[2] = {nullptr, (float*)d_out};
    unsigned short* layer_outb[2] = {H1bf, nullptr};

    for (int L = 0; L < 2; ++L) {
        long long waves = (NPAD / 32) * (FDIM / 64);
        gemm_bf16_k<<<cdiv(waves * 32, 256), 256, 0, stream>>>(layer_in[L], layer_W[L], nullptr,
                                                               feat, nullptr, NPAD, FDIM, FDIM);
        attn_scores_k<<<cdiv(NNODES * NHEAD, 256), 256, 0, stream>>>(feat, layer_al[L],
                                                                     layer_ar[L], el, er);
        hipMemsetAsync(menc, 0, (size_t)NNODES * NHEAD * 4, stream);
        hipMemsetAsync(den, 0, (size_t)NNODES * NHEAD * 4, stream);
        hipMemsetAsync(agg, 0, (size_t)NNODES * FDIM * 4, stream);
        edge_logits_k<<<cdiv((long long)NEDGES * NHEAD, 256), 256, 0, stream>>>(el, er, src, dst,
                                                                                ebuf, menc);
        decode_m_k<<<cdiv(NNODES * NHEAD, 256), 256, 0, stream>>>(menc, m);
        edge_exp_k<<<cdiv((long long)NEDGES * NHEAD, 256), 256, 0, stream>>>(ebuf, m, dst, den);
        edge_agg_k<<<cdiv((long long)NEDGES * 32, 256), 256, 0, stream>>>(ebuf, den, src, dst,
                                                                          feat, agg);
        finalize_k<<<cdiv((long long)NNODES * FDIM, 256), 256, 0, stream>>>(agg, layer_b[L],
                                                                            layer_outf[L],
                                                                            layer_outb[L]);
    }
    (void)in_sizes; (void)n_in; (void)out_size; (void)ws_size;
}
